// Convolution_49125835932146
// MI455X (gfx1250) — matrix operand order repaired and checked by reading, not compiled, not run on hardware
//
#include <hip/hip_runtime.h>

typedef __attribute__((ext_vector_type(16))) __bf16 v16bf;
typedef __attribute__((ext_vector_type(8)))  float  v8f;

static __device__ __forceinline__ unsigned short f32_to_bf16(float f) {
    unsigned int u = __float_as_uint(f);
    unsigned int rnd = 0x7FFFu + ((u >> 16) & 1u);   // round-to-nearest-even
    return (unsigned short)((u + rnd) >> 16);
}

// ---------------------------------------------------------------------------
// Phase A: build the equivariant 5x5x5 conv kernel and pack it as bf16 WMMA
// A-fragments. One block per tap. Fragment layout per tap:
//   frag f = kc*4 + Mblk (kc: K-chunk of 32 input channels, Mblk: 16 out ch)
//   within a fragment: lane*16 + j, following the CDNA5 16-bit A 16x32 layout:
//     lane<16 : M=lane,    j<8 -> K=j,    j>=8 -> K=j+8   (K 0..7, 16..23)
//     lane>=16: M=lane-16, j<8 -> K=j+8,  j>=8 -> K=j+16  (K 8..15, 24..31)
// ---------------------------------------------------------------------------
__global__ void build_wmma_kernel(const float* __restrict__ tpw,
                                  unsigned short* __restrict__ afrag) {
    const int tap = blockIdx.x;            // tap = dz*25 + dy*5 + dx
    const int a5 = tap / 25, b5 = (tap / 5) % 5, c5 = tap % 5;

    __shared__ float wrow[1280];           // (emb @ tp_weight) * cos(pi d)/5^1.5

    // lattice point and spherical harmonics (component normalization)
    const float lx = 0.5f * (float)(a5 - 2);
    const float ly = 0.5f * (float)(b5 - 2);
    const float lz = 0.5f * (float)(c5 - 2);
    const float d  = sqrtf(lx*lx + ly*ly + lz*lz);
    const float iv = 1.0f / fmaxf(d, 1e-12f);
    const float vx = lx*iv, vy = ly*iv, vz = lz*iv;
    const float r2 = vx*vx + vy*vy + vz*vz;

    const float s3  = 1.73205081f;
    const float s5  = 2.23606798f;
    const float s15 = 3.87298335f;

    float y1[3];                            // sh[1..3] = (s3 y, s3 z, s3 x)
    y1[0] = s3 * vy; y1[1] = s3 * vz; y1[2] = s3 * vx;
    float y2[5];                            // sh[4..8]
    y2[0] = s15 * vx * vy;
    y2[1] = s15 * vy * vz;
    y2[2] = 0.5f * s5 * (3.0f*vz*vz - r2);
    y2[3] = s15 * vx * vz;
    y2[4] = 0.5f * s15 * (vx*vx - vy*vy);

    // radial embedding -> wrow (cooperative)
    const float cfac = cosf(3.14159265358979f * d) / 11.1803398875f; // 5^1.5
    float e[5];
    #pragma unroll
    for (int j = 0; j < 5; ++j) {
        float t = (d - 0.25f * (float)j) * 4.0f;   // sigma = 1/4
        e[j] = expf(-t*t) / 1.12f;
    }
    for (int p = threadIdx.x; p < 1280; p += blockDim.x) {
        float s = 0.0f;
        #pragma unroll
        for (int j = 0; j < 5; ++j) s += e[j] * tpw[j*1280 + p];
        wrow[p] = s * cfac;
    }
    __syncthreads();

    // analytic Wigner-3j contraction constants (unit Frobenius, max-entry > 0)
    const float f0 = 0.176776695f;          // sqrt(1/32)
    const float f1 = 0.25f;                 // sqrt(1/16)
    const float i3 = 0.577350269f;          // 1/sqrt(3)
    const float q  = 0.316227766f;          // 1/sqrt(10)
    const float pq = 0.365148372f;          // sqrt(2/15)
    const float hq = 0.182574186f;          // 0.5*sqrt(2/15)

    // S[a][c] = sum_m C121[a][m][c] * y2[m], basis order (y,z,x)
    float S[3][3];
    S[0][0] = -hq*y2[2] - q*y2[4];
    S[0][1] =  q*y2[1];   S[1][0] = S[0][1];
    S[0][2] =  q*y2[0];   S[2][0] = S[0][2];
    S[1][1] =  pq*y2[2];
    S[1][2] =  q*y2[3];   S[2][1] = S[1][2];
    S[2][2] = -hq*y2[2] + q*y2[4];

    // 4096 bf16 elements per tap, 16 per thread
    for (int ei = 0; ei < 16; ++ei) {
        const int sub  = threadIdx.x * 16 + ei;   // 0..4095
        const int f    = sub >> 9;                // fragment 0..7
        const int kc   = f >> 2;
        const int Mblk = f & 3;
        const int rem  = sub & 511;
        const int lane = rem >> 4;
        const int j    = rem & 15;

        const int Ml = (lane < 16) ? lane : (lane - 16);
        const int Kl = (lane < 16) ? ((j < 8) ? j : j + 8)
                                   : ((j < 8) ? j + 8 : j + 16);
        const int o  = Mblk * 16 + Ml;            // output channel
        const int ci = kc * 32 + Kl;              // input channel

        float val;
        if (o < 16) {
            if (ci < 16) {
                val = f0 * wrow[0*256 + o*16 + ci];                    // b00
            } else {
                const int w = (ci - 16) / 3, c = (ci - 16) % 3;
                val = f1 * i3 * y1[c] * wrow[1*256 + o*16 + w];        // b01
            }
        } else {
            const int u = (o - 16) / 3, a = (o - 16) % 3;
            if (ci < 16) {
                val = f0 * i3 * y1[a] * wrow[3*256 + u*16 + ci];       // b10
            } else {
                const int w = (ci - 16) / 3, c = (ci - 16) % 3;
                val = f1 * S[a][c] * wrow[4*256 + u*16 + w];           // b11b
                if (a == c) val += f1 * i3 * wrow[2*256 + u*16 + w];   // b11a
            }
        }
        afrag[tap * 4096 + sub] = f32_to_bf16(val);
    }
}

// ---------------------------------------------------------------------------
// Phase B: implicit-GEMM 3D conv via v_wmma_f32_16x16x32_bf16 + fused si.
// One block per (batch, z, y): out tile M=64 (Cout) x N=32 (x-row), K=8000.
// 8 waves, each owning one (Mblk, Nblk) 16x16 tile / one v8f accumulator.
// ---------------------------------------------------------------------------
__global__ __launch_bounds__(256)
void conv_si_kernel(const float* __restrict__ x,
                    const float* __restrict__ wl0,
                    const float* __restrict__ wl1,
                    const unsigned short* __restrict__ afrag,
                    float* __restrict__ out) {
    const int bi   = blockIdx.x;
    const int b    = bi >> 10;
    const int zc   = (bi >> 5) & 31;
    const int yc   = bi & 31;
    const int tid  = threadIdx.x;
    const int lane = tid & 31;
    const int wv   = tid >> 5;
    const int Mblk = wv >> 1;
    const int Nblk = wv & 1;

    // pre-shifted bf16 stage: xs[(ci*5 + dx)*32 + n]  (20 KB)
    __shared__ __align__(32) unsigned short xs[64 * 5 * 32];

    v8f acc = {};

    const float* xb = x + (size_t)b * 64 * 32768;

    for (int s = 0; s < 25; ++s) {
        const int dz = s / 5, dy = s % 5;
        const int zr = zc + dz - 2, yr = yc + dy - 2;
        const bool rowOK = ((unsigned)zr < 32u) && ((unsigned)yr < 32u);

        __syncthreads();
        for (int idx = tid; idx < 10240; idx += 256) {
            const int ci  = idx / 160;
            const int rem = idx - ci * 160;
            const int dx  = rem >> 5;
            const int n   = rem & 31;
            const int xr  = n + dx - 2;
            float v = 0.0f;
            if (rowOK && (unsigned)xr < 32u)
                v = xb[ci * 32768 + zr * 1024 + yr * 32 + xr];
            xs[idx] = f32_to_bf16(v);
        }
        __syncthreads();

        const int tapbase = s * 5;
        #pragma unroll
        for (int dx = 0; dx < 5; ++dx) {
            const unsigned short* ap0 =
                afrag + (size_t)(tapbase + dx) * 4096 + Mblk * 512 + lane * 16;
            __builtin_prefetch(ap0 + 4096, 0, 1);   // next tap's fragment
            #pragma unroll
            for (int kc = 0; kc < 2; ++kc) {
                // A fragment: 32B aligned per-lane vector (layout from phase A)
                v16bf a = *(const v16bf*)(ap0 + kc * 2048);
                // B fragment: lane -> n = lane&15, element s -> k in table order
                const int nb = Nblk * 16 + (lane & 15);
                const int hb = lane >> 4;
                v16bf bf;
                #pragma unroll
                for (int e = 0; e < 16; ++e) {
                    const int kk = (e < 8) ? (8 * hb + e) : (16 + 8 * hb + (e - 8));
                    const int ci = kc * 32 + kk;
                    bf[e] = __builtin_bit_cast(__bf16, xs[(ci * 5 + dx) * 32 + nb]);
                }
                acc = __builtin_amdgcn_wmma_f32_16x16x32_bf16(
                        false, a, false, bf, (short)0, acc, false, false);
            }
        }
    }

    // ---- epilogue: restage center row as f32, fuse si, write out ----------
    __syncthreads();
    float* xc = reinterpret_cast<float*>(xs);     // 2048 f32 = 8 KB, fits
    for (int idx = tid; idx < 2048; idx += 256) {
        const int ci = idx >> 5, n = idx & 31;
        xc[idx] = xb[ci * 32768 + zc * 1024 + yc * 32 + n];
    }
    __syncthreads();

    // C/D 16x16 f32 layout: VGPR r -> M = r + 8*(lane>=16), N = lane&15
    const int n     = Nblk * 16 + (lane & 15);
    const int rbase = Mblk * 16 + ((lane >> 4) << 3);
    float* ob = out + (size_t)b * 64 * 32768 + zc * 1024 + yc * 32 + n;

    #pragma unroll
    for (int r = 0; r < 8; ++r) {
        const int co = rbase + r;
        float si = 0.0f;
        if (co < 16) {
            #pragma unroll
            for (int u = 0; u < 16; ++u)
                si += wl0[u * 16 + co] * xc[u * 32 + n];
        } else {
            const int v = (co - 16) / 3, m = (co - 16) % 3;
            #pragma unroll
            for (int u = 0; u < 16; ++u)
                si += wl1[u * 16 + v] * xc[(16 + u * 3 + m) * 32 + n];
        }
        si *= 0.25f;                                  // 1/sqrt(mul)
        ob[(size_t)co * 32768] = si + 0.1f * acc[r];
    }
}

// ---------------------------------------------------------------------------
extern "C" void kernel_launch(void* const* d_in, const int* in_sizes, int n_in,
                              void* d_out, int out_size, void* d_ws, size_t ws_size,
                              hipStream_t stream) {
    const float* x   = (const float*)d_in[0];   // (4, 64, 32, 32, 32)
    const float* wl0 = (const float*)d_in[1];   // (16, 16)
    const float* wl1 = (const float*)d_in[2];   // (16, 16)
    const float* tpw = (const float*)d_in[3];   // (5, 1280)

    unsigned short* afrag = (unsigned short*)d_ws;  // 125*4096 bf16 = 1 MB

    build_wmma_kernel<<<125, 256, 0, stream>>>(tpw, afrag);
    conv_si_kernel<<<4096, 256, 0, stream>>>(x, wl0, wl1, afrag, (float*)d_out);
}
